// Attention_14817637171464
// MI455X (gfx1250) — matrix operand order repaired and checked by reading, not compiled, not run on hardware
//
#include <hip/hip_runtime.h>
#include <hip/hip_bf16.h>

// ---------------------------------------------------------------------------
// Causal multi-head attention for MI455X (gfx1250), wave32 + WMMA f16.
//   x:[2,2048,1024] -> LN -> QKV GEMM -> causal flash attention -> out GEMM
// All matmuls on v_wmma_f32_16x16x32_f16 (f16 in, f32 accumulate).
// GEMMs: 32x64 tile per wave, explicit two-generation ping-pong over K so the
// double buffer needs no register copies (no v_mov walls, no spills).
// ---------------------------------------------------------------------------

typedef __attribute__((ext_vector_type(16))) _Float16 v16h;
typedef __attribute__((ext_vector_type(8)))  _Float16 v8h;
typedef __attribute__((ext_vector_type(8)))  float    v8f;

#define DIM    1024
#define SEQ    2048
#define BATCH  2
#define HEADS  16
#define DHEAD  64
#define INNER  1024
#define ROWS   (BATCH * SEQ)       // 4096
#define NEG_BIG (-3.0e38f)

// ---- WMMA helpers ---------------------------------------------------------

__device__ __forceinline__ v8f wmma_f16(v16h a, v16h b, v8f c) {
  // (neg_a, A, neg_b, B, c_mod, C, reuse_a, reuse_b)
  return __builtin_amdgcn_wmma_f32_16x16x32_f16(false, a, false, b,
                                                (short)0, c, false, false);
}

// A fragment (16x32 f16, MxK). ISA layout:
//  lanes 0-15 : row M=lane,    halves 0-7 -> K=0..7,  halves 8-15 -> K=16..23
//  lanes 16-31: row M=lane-16, halves 0-7 -> K=8..15, halves 8-15 -> K=24..31
template <typename PT>
__device__ __forceinline__ v16h load_a_frag(PT base, int ld) {
  int lane = threadIdx.x & 31;
  PT p = base + (size_t)(lane & 15) * ld + ((lane & 16) ? 8 : 0);
  v8h lo = *(const v8h*)(p);
  v8h hi = *(const v8h*)(p + 16);
  v16h a;
#pragma unroll
  for (int i = 0; i < 8; ++i) { a[i] = lo[i]; a[i + 8] = hi[i]; }
  return a;
}

// B fragment (32x16 f16, KxN) where column n of B is contiguous in memory,
// i.e. we pass a row-major matrix W[n][k] and the tile computes A x W^T.
//  lanes 0-15 : col N=lane,    halves 0-7 -> K=0..7,  halves 8-15 -> K=16..23
//  lanes 16-31: col N=lane-16, halves 0-7 -> K=8..15, halves 8-15 -> K=24..31
__device__ __forceinline__ v16h load_b_frag(const _Float16* base, int ld) {
  int lane = threadIdx.x & 31;
  const _Float16* p = base + (size_t)(lane & 15) * ld + ((lane & 16) ? 8 : 0);
  v8h lo = *(const v8h*)(p);
  v8h hi = *(const v8h*)(p + 16);
  v16h b;
#pragma unroll
  for (int i = 0; i < 8; ++i) { b[i] = lo[i]; b[i + 8] = hi[i]; }
  return b;
}

// C/D f32 16x16 layout: element (M,N) lives in lane = (N&15) + 16*(M>=8),
// vgpr r = M%8.  Per-lane: N = lane&15, Mbase = (lane&16)?8:0.

// ---- Kernel 1: LayerNorm -> f16 -------------------------------------------

__global__ void ln_kernel(const float* __restrict__ x,
                          const float* __restrict__ gamma,
                          const float* __restrict__ beta,
                          _Float16* __restrict__ xn) {
  __shared__ float red_s[8];
  __shared__ float red_ss[8];
  const int row = blockIdx.x;
  const int t = threadIdx.x;                 // 256 threads, 4 elems each
  const float* xr = x + (size_t)row * DIM;
  float4 v = ((const float4*)xr)[t];
  float s  = v.x + v.y + v.z + v.w;
  float ss = v.x * v.x + v.y * v.y + v.z * v.z + v.w * v.w;
#pragma unroll
  for (int m = 16; m >= 1; m >>= 1) {
    s  += __shfl_xor(s,  m, 32);
    ss += __shfl_xor(ss, m, 32);
  }
  const int wid = t >> 5;
  if ((t & 31) == 0) { red_s[wid] = s; red_ss[wid] = ss; }
  __syncthreads();
  s = 0.0f; ss = 0.0f;
#pragma unroll
  for (int i = 0; i < 8; ++i) { s += red_s[i]; ss += red_ss[i]; }
  const float mu   = s * (1.0f / DIM);
  const float var  = ss * (1.0f / DIM) - mu * mu;
  const float rstd = rsqrtf(var + 1e-5f);
  float4 g = ((const float4*)gamma)[t];
  float4 b = ((const float4*)beta)[t];
  _Float16* o = xn + (size_t)row * DIM + t * 4;
  o[0] = (_Float16)((v.x - mu) * rstd * g.x + b.x);
  o[1] = (_Float16)((v.y - mu) * rstd * g.y + b.y);
  o[2] = (_Float16)((v.z - mu) * rstd * g.z + b.z);
  o[3] = (_Float16)((v.w - mu) * rstd * g.w + b.w);
}

// ---- Kernel 2: f32 -> f16 weight conversion -------------------------------

__global__ void cvt_f32_f16(const float* __restrict__ in,
                            _Float16* __restrict__ out, int n) {
  int i = blockIdx.x * blockDim.x + threadIdx.x;
  int stride = gridDim.x * blockDim.x;
  for (; i < n; i += stride) out[i] = (_Float16)in[i];
}

// ---- shared GEMM core: 32x64 tile, two-generation ping-pong over K --------

struct FragSet { v16h a0, a1, b0, b1, b2, b3; };

__device__ __forceinline__ void load_set(FragSet& f, const _Float16* A0,
                                         const _Float16* A1, const _Float16* Wn,
                                         int k, int ld) {
  f.a0 = load_a_frag(A0 + k, ld);
  f.a1 = load_a_frag(A1 + k, ld);
  f.b0 = load_b_frag(Wn + (size_t)0  * ld + k, ld);
  f.b1 = load_b_frag(Wn + (size_t)16 * ld + k, ld);
  f.b2 = load_b_frag(Wn + (size_t)32 * ld + k, ld);
  f.b3 = load_b_frag(Wn + (size_t)48 * ld + k, ld);
}

__device__ __forceinline__ void step_set(const FragSet& f, v8f acc[2][4]) {
  acc[0][0] = wmma_f16(f.a0, f.b0, acc[0][0]);
  acc[1][0] = wmma_f16(f.a1, f.b0, acc[1][0]);
  acc[0][1] = wmma_f16(f.a0, f.b1, acc[0][1]);
  acc[1][1] = wmma_f16(f.a1, f.b1, acc[1][1]);
  acc[0][2] = wmma_f16(f.a0, f.b2, acc[0][2]);
  acc[1][2] = wmma_f16(f.a1, f.b2, acc[1][2]);
  acc[0][3] = wmma_f16(f.a0, f.b3, acc[0][3]);
  acc[1][3] = wmma_f16(f.a1, f.b3, acc[1][3]);
}

// Computes acc[2][4] (32x64 tile at rows m0, cols n0) of  A(MxK) x W(NxK)^T.
__device__ __forceinline__ void gemm_tile_32x64(const _Float16* A, const _Float16* W,
                                                int m0, int n0, int ld,
                                                v8f acc[2][4]) {
  const _Float16* A0 = A + (size_t)m0 * ld;
  const _Float16* A1 = A + (size_t)(m0 + 16) * ld;
  const _Float16* Wn = W + (size_t)n0 * ld;
  FragSet fa, fb;
  load_set(fa, A0, A1, Wn, 0, ld);               // gen 0
  for (int k = 32; k <= ld - 96; k += 64) {
    load_set(fb, A0, A1, Wn, k, ld);             // odd gen
    step_set(fa, acc);                           // even gen
    load_set(fa, A0, A1, Wn, k + 32, ld);        // next even gen
    step_set(fb, acc);                           // odd gen
  }
  load_set(fb, A0, A1, Wn, ld - 32, ld);         // last odd gen
  step_set(fa, acc);                             // gen ld-64
  step_set(fb, acc);                             // gen ld-32
}

// ---- Kernel 3: QKV GEMM  qkv[m,e] = sum_k xn[m,k] * Wqkv[e,k] -------------
// Results scattered to:
//   Q  [bh, n, d]  (pre-scaled by 1/8)
//   K  [bh, n, d]
//   VT [bh, d, n]  (transposed so PV B-fragments are contiguous)

__global__ void __launch_bounds__(128, 1)
qkv_gemm(const _Float16* __restrict__ A,
         const _Float16* __restrict__ W,
         _Float16* __restrict__ Q,
         _Float16* __restrict__ K,
         _Float16* __restrict__ VT) {
  const int wid  = threadIdx.x >> 5;
  const int lane = threadIdx.x & 31;
  const int job  = blockIdx.x * 4 + wid;
  const int m0   = (job % (ROWS / 32)) * 32;   // 128 row tiles
  const int n0   = (job / (ROWS / 32)) * 64;   // 48 column strips

  v8f acc[2][4] = {};
  gemm_tile_32x64(A, W, m0, n0, DIM, acc);

  const int N  = lane & 15;
  const int Mb = (lane & 16) ? 8 : 0;
#pragma unroll
  for (int mi = 0; mi < 2; ++mi) {
#pragma unroll
    for (int t = 0; t < 4; ++t) {
      const int e     = n0 + 16 * t + N;
      const int which = e >> 10;             // 0=q 1=k 2=v
      const int hd    = e & 1023;
      const int h     = hd >> 6;
      const int d     = hd & 63;
#pragma unroll
      for (int r = 0; r < 8; ++r) {
        const int m  = m0 + 16 * mi + Mb + r;
        const int bb = m >> 11;              // /SEQ
        const int n  = m & (SEQ - 1);
        const int bh = bb * HEADS + h;
        const float val = acc[mi][t][r];
        if (which == 0)
          Q[((size_t)bh * SEQ + n) * DHEAD + d] = (_Float16)(val * 0.125f);
        else if (which == 1)
          K[((size_t)bh * SEQ + n) * DHEAD + d] = (_Float16)val;
        else
          VT[((size_t)bh * DHEAD + d) * SEQ + n] = (_Float16)val;
      }
    }
  }
}

// ---- Kernel 4: causal flash attention, one wave per (bh, 16-row block) ----

__global__ void __launch_bounds__(128, 1)
attn_kernel(const _Float16* __restrict__ Q,
            const _Float16* __restrict__ K,
            const _Float16* __restrict__ VT,
            _Float16* __restrict__ O) {
  __shared__ __align__(32) _Float16 plds[4][16 * 32];   // P relayout buffer
  const int wid  = threadIdx.x >> 5;
  const int lane = threadIdx.x & 31;
  const int job  = blockIdx.x * 4 + wid;
  const int bh   = job >> 7;                 // / (SEQ/16)
  const int i0   = (job & 127) * 16;
  const _Float16* Qb = Q  + (size_t)bh * SEQ * DHEAD;
  const _Float16* Kb = K  + (size_t)bh * SEQ * DHEAD;
  const _Float16* Vb = VT + (size_t)bh * DHEAD * SEQ;
  _Float16* pw = &plds[wid][0];

  // Q fragments for the two K-steps of the d=64 reduction.
  const v16h aq0 = load_a_frag(Qb + (size_t)i0 * DHEAD + 0,  DHEAD);
  const v16h aq1 = load_a_frag(Qb + (size_t)i0 * DHEAD + 32, DHEAD);

  v8f acc[4] = {};                // 16 x 64 output accumulator
  float mrow[8], lrow[8];
#pragma unroll
  for (int r = 0; r < 8; ++r) { mrow[r] = NEG_BIG; lrow[r] = 0.0f; }

  const int N  = lane & 15;
  const int Mb = (lane & 16) ? 8 : 0;

  for (int j = 0; j <= i0 + 15; j += 32) {
    // ---- S = Q K^T for columns j..j+31 (two 16x16 f32 tiles) ----
    v8f S0 = {}, S1 = {};
    {
      v16h bk0 = load_b_frag(Kb + (size_t)j * DHEAD + 0,  DHEAD);
      v16h bk1 = load_b_frag(Kb + (size_t)j * DHEAD + 32, DHEAD);
      v16h bk2 = load_b_frag(Kb + (size_t)(j + 16) * DHEAD + 0,  DHEAD);
      v16h bk3 = load_b_frag(Kb + (size_t)(j + 16) * DHEAD + 32, DHEAD);
      S0 = wmma_f16(aq0, bk0, S0);
      S0 = wmma_f16(aq1, bk1, S0);
      S1 = wmma_f16(aq0, bk2, S1);
      S1 = wmma_f16(aq1, bk3, S1);
    }
    // V B-fragments do not depend on P: issue the loads now so they overlap
    // the softmax VALU work and the LDS round-trip.
    v16h bv[4];
#pragma unroll
    for (int t = 0; t < 4; ++t)
      bv[t] = load_b_frag(Vb + (size_t)(t * 16) * SEQ + j, SEQ);

    // ---- causal mask ----
    if (j + 31 > i0) {
#pragma unroll
      for (int r = 0; r < 8; ++r) {
        const int row = i0 + Mb + r;
        if (j + N      > row) S0[r] = NEG_BIG;
        if (j + 16 + N > row) S1[r] = NEG_BIG;
      }
    }
    // ---- online softmax (row stats align with C-layout rows) ----
    float alpha[8];
#pragma unroll
    for (int r = 0; r < 8; ++r) {
      float mc = fmaxf(S0[r], S1[r]);
#pragma unroll
      for (int msk = 8; msk >= 1; msk >>= 1) mc = fmaxf(mc, __shfl_xor(mc, msk, 32));
      const float mn = fmaxf(mrow[r], mc);
      const float p0 = __expf(S0[r] - mn);
      const float p1 = __expf(S1[r] - mn);
      S0[r] = p0; S1[r] = p1;
      float rs = p0 + p1;
#pragma unroll
      for (int msk = 8; msk >= 1; msk >>= 1) rs += __shfl_xor(rs, msk, 32);
      alpha[r] = __expf(mrow[r] - mn);
      mrow[r]  = mn;
      lrow[r]  = lrow[r] * alpha[r] + rs;
    }
#pragma unroll
    for (int t = 0; t < 4; ++t)
#pragma unroll
      for (int r = 0; r < 8; ++r) acc[t][r] = acc[t][r] * alpha[r];

    // ---- P: C-layout -> LDS (16x32 row-major) -> A-fragment ----
#pragma unroll
    for (int r = 0; r < 8; ++r) {
      pw[(Mb + r) * 32 + N]      = (_Float16)S0[r];
      pw[(Mb + r) * 32 + 16 + N] = (_Float16)S1[r];
    }
    asm volatile("s_wait_dscnt 0x0" ::: "memory");   // wave-local LDS RAW
    v16h pa = load_a_frag((const _Float16*)pw, 32);

    // ---- acc += P x V ----
#pragma unroll
    for (int t = 0; t < 4; ++t) acc[t] = wmma_f16(pa, bv[t], acc[t]);
  }

  // ---- normalize and store to [b, n, h, d] (== [b, n, inner]) ----
  const int b = bh >> 4;
  const int h = bh & (HEADS - 1);
#pragma unroll
  for (int r = 0; r < 8; ++r) {
    const float rl = 1.0f / lrow[r];
    const int n = i0 + Mb + r;
    const size_t base = (((size_t)b * SEQ + n) * HEADS + h) * DHEAD;
#pragma unroll
    for (int t = 0; t < 4; ++t)
      O[base + t * 16 + N] = (_Float16)(acc[t][r] * rl);
  }
}

// ---- Kernel 5: output GEMM  out[m,d] = sum_e ao[m,e] * Wout[d,e] + b[d] ---

__global__ void __launch_bounds__(128, 1)
out_gemm(const _Float16* __restrict__ A,
         const _Float16* __restrict__ W,
         const float* __restrict__ bias,
         float* __restrict__ out) {
  const int wid  = threadIdx.x >> 5;
  const int lane = threadIdx.x & 31;
  const int job  = blockIdx.x * 4 + wid;
  const int m0   = (job % (ROWS / 32)) * 32;   // 128 row tiles
  const int n0   = (job / (ROWS / 32)) * 64;   // 16 column strips

  v8f acc[2][4] = {};
  gemm_tile_32x64(A, W, m0, n0, INNER, acc);

  const int N  = lane & 15;
  const int Mb = (lane & 16) ? 8 : 0;
#pragma unroll
  for (int mi = 0; mi < 2; ++mi) {
#pragma unroll
    for (int t = 0; t < 4; ++t) {
      const int d = n0 + 16 * t + N;
      const float bv = bias[d];
#pragma unroll
      for (int r = 0; r < 8; ++r)
        out[(size_t)(m0 + 16 * mi + Mb + r) * DIM + d] = acc[mi][t][r] + bv;
    }
  }
}

// ---- host launch ----------------------------------------------------------

extern "C" void kernel_launch(void* const* d_in, const int* in_sizes, int n_in,
                              void* d_out, int out_size, void* d_ws, size_t ws_size,
                              hipStream_t stream) {
  const float* x     = (const float*)d_in[0];
  const float* gamma = (const float*)d_in[1];
  const float* beta  = (const float*)d_in[2];
  const float* wqkv  = (const float*)d_in[3];
  const float* wout  = (const float*)d_in[4];
  const float* bout  = (const float*)d_in[5];
  float* out = (float*)d_out;

  // Workspace carve-up (f16 elements). Total ~42 MB.
  _Float16* ws = (_Float16*)d_ws;
  size_t o = 0;
  _Float16* xnh = ws + o;  o += (size_t)ROWS * DIM;          // LN output (reused as attn out)
  _Float16* wqh = ws + o;  o += (size_t)3 * INNER * DIM;     // Wqkv f16
  _Float16* woh = ws + o;  o += (size_t)DIM * INNER;         // Wout f16
  _Float16* Qh  = ws + o;  o += (size_t)BATCH * HEADS * SEQ * DHEAD;
  _Float16* Kh  = ws + o;  o += (size_t)BATCH * HEADS * SEQ * DHEAD;
  _Float16* VTh = ws + o;  o += (size_t)BATCH * HEADS * SEQ * DHEAD;
  _Float16* Oh  = xnh;     // xn is dead after qkv_gemm; alias for attn output

  ln_kernel<<<ROWS, 256, 0, stream>>>(x, gamma, beta, xnh);
  cvt_f32_f16<<<1536, 256, 0, stream>>>(wqkv, wqh, 3 * INNER * DIM);
  cvt_f32_f16<<<512, 256, 0, stream>>>(wout, woh, DIM * INNER);
  qkv_gemm<<<(128 * 48) / 4, 128, 0, stream>>>(xnh, wqh, Qh, Kh, VTh);
  attn_kernel<<<(BATCH * HEADS * (SEQ / 16)) / 4, 128, 0, stream>>>(Qh, Kh, VTh, Oh);
  out_gemm<<<(128 * 16) / 4, 128, 0, stream>>>(Oh, woh, bout, out);
}
